// MultiHeadAttentionLayer_27779848470630
// MI455X (gfx1250) — matrix operand order repaired and checked by reading, not compiled, not run on hardware
//
#include <hip/hip_runtime.h>

typedef __attribute__((ext_vector_type(16))) _Float16 v16h;
typedef __attribute__((ext_vector_type(8)))  float    v8f;

#define HD      64   // NUM_HEADS * OUT_DIM
#define IN_DIM  64
#define NHEADS  4
#define DHEAD   16

// ---- A-tile loader: 16 rows x 32 K of f16, contiguous K order (matches packed B) ----
// lane<16 : row=lane,    elems 0..15 -> K=k0+0..15
// lane>=16: row=lane-16, elems 0..15 -> K=k0+16..31
__device__ __forceinline__ v16h load_a_tile(const float* __restrict__ src,
                                            int row0, int ld, int k0, int lane) {
    int m  = lane & 15;
    int kb = (lane >> 4) << 4;                    // 0 or 16
    const float4* r = (const float4*)(src + (size_t)(row0 + m) * ld + k0 + kb);
    float4 p0 = r[0], p1 = r[1];                  // K = kb+0..7
    float4 p2 = r[2], p3 = r[3];                  // K = kb+8..15
    v16h a;
    a[0]=(_Float16)p0.x; a[1]=(_Float16)p0.y; a[2]=(_Float16)p0.z; a[3]=(_Float16)p0.w;
    a[4]=(_Float16)p1.x; a[5]=(_Float16)p1.y; a[6]=(_Float16)p1.z; a[7]=(_Float16)p1.w;
    a[8]=(_Float16)p2.x; a[9]=(_Float16)p2.y; a[10]=(_Float16)p2.z; a[11]=(_Float16)p2.w;
    a[12]=(_Float16)p3.x; a[13]=(_Float16)p3.y; a[14]=(_Float16)p3.z; a[15]=(_Float16)p3.w;
    return a;
}

// Packed-B index: P[((mtx*2 + slab)*4 + tile)*32 + lane], 32 bytes per lane
__device__ __forceinline__ size_t packedB_idx(int mtx, int slab, int tile, int lane) {
    return (size_t)(((mtx * 2 + slab) * 4 + tile) * 32 + lane);
}

// ===================== kernel P: prepack weights into WMMA B layout (f16) =====================
// 128 threads = 4 waves; wave handles one matrix. B layout, contiguous K (same as A loader):
// lane holds col n = tile*16+(lane&15); elem j -> K = slab*32 + 16*(lane>=16) + j
__global__ __launch_bounds__(128) void pack_weights_kernel(
    const float* __restrict__ W0, const float* __restrict__ W1,
    const float* __restrict__ W2, const float* __restrict__ W3,
    v16h* __restrict__ P)
{
    int lane = threadIdx.x & 31;
    int mtx  = threadIdx.x >> 5;
    const float* Ws[4] = {W0, W1, W2, W3};
    const float* W = Ws[mtx];
#pragma unroll
    for (int slab = 0; slab < 2; ++slab) {
#pragma unroll
        for (int tile = 0; tile < 4; ++tile) {
            int n  = tile * 16 + (lane & 15);
            int kb = slab * 32 + ((lane >> 4) << 4);
            v16h b;
#pragma unroll
            for (int j = 0; j < 16; ++j)
                b[j] = (_Float16)W[(size_t)(kb + j) * HD + n];
            P[packedB_idx(mtx, slab, tile, lane)] = b;
        }
    }
}

// ===================== kernel 0: zero accumulators =====================
__global__ void zero_kernel(float* __restrict__ p, size_t n) {
    size_t i = (size_t)blockIdx.x * blockDim.x + threadIdx.x;
    if (i < n) p[i] = 0.0f;
}

// ===================== kernel 1: node projections Q,K,V =====================
// grid = n_nodes/16 blocks, 128 threads = 4 waves; wave w -> output cols [16w,16w+16)
__global__ __launch_bounds__(128) void node_proj_kernel(
    const float* __restrict__ x,
    const v16h*  __restrict__ P,              // packed WQ(0), WK(1), WV(2)
    const float* __restrict__ bQ, const float* __restrict__ bK, const float* __restrict__ bV,
    float* __restrict__ Qp, float* __restrict__ Kp, float* __restrict__ Vp,
    int n_nodes)
{
    int lane = threadIdx.x & 31;
    int wave = threadIdx.x >> 5;
    int row0 = blockIdx.x * 16;
    if (row0 >= n_nodes) return;

    v16h a0 = load_a_tile(x, row0, IN_DIM, 0,  lane);
    v16h a1 = load_a_tile(x, row0, IN_DIM, 32, lane);

    const float* bs[3]   = {bQ, bK, bV};
    float*       outs[3] = {Qp, Kp, Vp};

#pragma unroll
    for (int mtx = 0; mtx < 3; ++mtx) {
        v16h b0 = P[packedB_idx(mtx, 0, wave, lane)];   // 2x global_load_b128
        v16h b1 = P[packedB_idx(mtx, 1, wave, lane)];
        v8f c = {};
        c = __builtin_amdgcn_wmma_f32_16x16x32_f16(false, a0, false, b0, (short)0, c, false, false);
        c = __builtin_amdgcn_wmma_f32_16x16x32_f16(false, a1, false, b1, (short)0, c, false, false);
        int   n    = wave * 16 + (lane & 15);
        float bias = bs[mtx][n];
        int   mb   = row0 + ((lane >> 4) << 3);         // D rows: lane half selects +8
#pragma unroll
        for (int r = 0; r < 8; ++r)
            outs[mtx][(size_t)(mb + r) * HD + n] = c[r] + bias;
    }
}

// ===================== kernel 2: fused edge projection + attention scatter =====================
// 256 threads = 8 waves; each wave owns 16 edges: prefetch gathers, WMMA-project E into
// its private LDS slab, then each lane handles two (edge, head) pairs and scatters.
__global__ __launch_bounds__(256) void edge_attn_kernel(
    const float* __restrict__ edge_attr,
    const int*   __restrict__ edge_index, int n_edges,
    const v16h*  __restrict__ P,              // packed WE at mtx=3
    const float* __restrict__ bE,
    const float* __restrict__ Qp, const float* __restrict__ Kp, const float* __restrict__ Vp,
    float* __restrict__ wV, float* __restrict__ Z)
{
    __shared__ float Elds[8][16][HD + 4];     // +4 floats pad vs 64-bank LDS
    int lane = threadIdx.x & 31;
    int wave = threadIdx.x >> 5;
    int e0   = (blockIdx.x * 8 + wave) * 16;
    if (e0 >= n_edges) return;

    // ---- resolve gather addresses early; prefetch rows under the matmul ----
    int eloc = lane & 15;                      // both head-pairs of a lane share the edge
    int e    = e0 + eloc;
    int src  = edge_index[e];
    int dst  = edge_index[n_edges + e];
    const float* Kb = Kp + (size_t)src * HD;   // 256 B rows = 2 cachelines each
    const float* Qb = Qp + (size_t)dst * HD;
    const float* Vb = Vp + (size_t)src * HD;
    __builtin_prefetch(Kb, 0, 3);  __builtin_prefetch(Kb + 32, 0, 3);
    __builtin_prefetch(Qb, 0, 3);  __builtin_prefetch(Qb + 32, 0, 3);
    __builtin_prefetch(Vb, 0, 3);  __builtin_prefetch(Vb + 32, 0, 3);

    // ---- E = edge_attr @ WE + bE via WMMA, into this wave's LDS slab ----
    v16h a0 = load_a_tile(edge_attr, e0, IN_DIM, 0,  lane);
    v16h a1 = load_a_tile(edge_attr, e0, IN_DIM, 32, lane);

#pragma unroll
    for (int t = 0; t < 4; ++t) {
        v16h b0 = P[packedB_idx(3, 0, t, lane)];
        v16h b1 = P[packedB_idx(3, 1, t, lane)];
        v8f c = {};
        c = __builtin_amdgcn_wmma_f32_16x16x32_f16(false, a0, false, b0, (short)0, c, false, false);
        c = __builtin_amdgcn_wmma_f32_16x16x32_f16(false, a1, false, b1, (short)0, c, false, false);
        int   n    = t * 16 + (lane & 15);
        float bias = bE[n];
        int   mb   = (lane >> 4) << 3;
#pragma unroll
        for (int r = 0; r < 8; ++r)
            Elds[wave][mb + r][n] = c[r] + bias;  // same-wave producer/consumer: no barrier
    }

    // ---- two (edge, head) pairs per lane: heads h and h+2 ----
#pragma unroll
    for (int p = 0; p < 2; ++p) {
        int h = (lane >> 4) + 2 * p;
        const float4* Ks = (const float4*)(Kb + h * DHEAD);
        const float4* Qd = (const float4*)(Qb + h * DHEAD);
        const float4* Vs = (const float4*)(Vb + h * DHEAD);
        const float4* Ee = (const float4*)(&Elds[wave][eloc][h * DHEAD]);

        float s = 0.0f;
#pragma unroll
        for (int q = 0; q < 4; ++q) {
            float4 kv = Ks[q], qv = Qd[q], ev = Ee[q];
            s += kv.x * qv.x * ev.x + kv.y * qv.y * ev.y
               + kv.z * qv.z * ev.z + kv.w * qv.w * ev.w;
        }
        s *= 0.25f;                                // 1/sqrt(DHEAD)
        s = fminf(5.0f, fmaxf(-5.0f, s));
        s = __expf(s);

        float* outrow = wV + (size_t)dst * HD + h * DHEAD;
#pragma unroll
        for (int q = 0; q < 4; ++q) {
            float4 vv = Vs[q];
            atomicAdd(&outrow[q * 4 + 0], vv.x * s);
            atomicAdd(&outrow[q * 4 + 1], vv.y * s);
            atomicAdd(&outrow[q * 4 + 2], vv.z * s);
            atomicAdd(&outrow[q * 4 + 3], vv.w * s);
        }
        atomicAdd(&Z[dst * NHEADS + h], s);
    }
}

// ===================== kernel 3: normalize =====================
__global__ void finalize_kernel(const float* __restrict__ wV, const float* __restrict__ Z,
                                float* __restrict__ out, int n_nodes) {
    int i = blockIdx.x * blockDim.x + threadIdx.x;
    if (i < n_nodes * HD) {
        int node = i >> 6;
        int h    = (i & 63) >> 4;
        out[i] = wV[i] / (Z[node * NHEADS + h] + 1e-6f);
    }
}

extern "C" void kernel_launch(void* const* d_in, const int* in_sizes, int n_in,
                              void* d_out, int out_size, void* d_ws, size_t ws_size,
                              hipStream_t stream) {
    const float* x         = (const float*)d_in[0];
    const float* edge_attr = (const float*)d_in[1];
    const int*   edge_idx  = (const int*)  d_in[2];
    const float* WQ = (const float*)d_in[3];
    const float* bQ = (const float*)d_in[4];
    const float* WK = (const float*)d_in[5];
    const float* bK = (const float*)d_in[6];
    const float* WE = (const float*)d_in[7];
    const float* bE = (const float*)d_in[8];
    const float* WV = (const float*)d_in[9];
    const float* bV = (const float*)d_in[10];
    float* out = (float*)d_out;

    int n_nodes = in_sizes[0] / IN_DIM;
    int n_edges = in_sizes[2] / 2;

    // workspace layout (f32): Q | K | V | wV | Z | packedB(f16)
    float* Qp = (float*)d_ws;
    float* Kp = Qp + (size_t)n_nodes * HD;
    float* Vp = Kp + (size_t)n_nodes * HD;
    float* wV = Vp + (size_t)n_nodes * HD;
    float* Zs = wV + (size_t)n_nodes * HD;
    v16h*  Pb = (v16h*)(Zs + (size_t)n_nodes * NHEADS);   // 4 mtx * 8 tiles * 32 lanes * 32B = 32 KB

    // pack order: 0=WQ, 1=WK, 2=WV, 3=WE
    pack_weights_kernel<<<1, 128, 0, stream>>>(WQ, WK, WV, WE, Pb);

    size_t nz = (size_t)n_nodes * HD + (size_t)n_nodes * NHEADS;  // wV and Z contiguous
    zero_kernel<<<(unsigned)((nz + 255) / 256), 256, 0, stream>>>(wV, nz);

    node_proj_kernel<<<(n_nodes + 15) / 16, 128, 0, stream>>>(
        x, Pb, bQ, bK, bV, Qp, Kp, Vp, n_nodes);

    int edge_waves  = (n_edges + 15) / 16;
    int edge_blocks = (edge_waves + 7) / 8;
    edge_attn_kernel<<<edge_blocks, 256, 0, stream>>>(
        edge_attr, edge_idx, n_edges, Pb, bE, Qp, Kp, Vp, wV, Zs);

    finalize_kernel<<<(n_nodes * HD + 255) / 256, 256, 0, stream>>>(wV, Zs, out, n_nodes);
}
